// LSTMClassifier_6588479832550
// MI455X (gfx1250) — matrix operand order repaired and checked by reading, not compiled, not run on hardware
//
#include <hip/hip_runtime.h>
#include <math.h>

// Problem constants (match reference)
#define B_    128
#define T_    1024
#define FIN   128
#define CC    64     // conv channels
#define TP    1022   // T' after VALID conv k=3
#define H_    512
#define G4    2048   // 4*H
#define OUTN  10

typedef __attribute__((ext_vector_type(16))) __bf16 v16bf;
typedef __attribute__((ext_vector_type(8)))  __bf16 v8bf;
typedef __attribute__((ext_vector_type(8)))  float  v8f;

// ---------------------------------------------------------------------------
// WMMA fragment helpers (CDNA5 wave32 layouts, CDNA5 ISA §7.12.2)
// ---------------------------------------------------------------------------
__device__ __forceinline__ v16bf make_frag(v8bf lo, v8bf hi) {
  v16bf r;
#pragma unroll
  for (int i = 0; i < 8; ++i) { r[i] = lo[i]; r[i + 8] = hi[i]; }
  return r;
}

// A fragment (16x32, row-major source):
//   lanes 0-15 : M=lane,    K = k0+{0..7} then k0+{16..23}
//   lanes16-31 : M=lane-16, K = k0+{8..15} then k0+{24..31}
__device__ __forceinline__ v16bf load_a(const __bf16* base, int ld, int m0, int k0) {
  int lane = threadIdx.x & 31;
  const __bf16* p = base + (size_t)(m0 + (lane & 15)) * ld + k0 + ((lane >> 4) << 3);
  v8bf lo = *(const v8bf*)p;
  v8bf hi = *(const v8bf*)(p + 16);
  return make_frag(lo, hi);
}

// B fragment (32x16, [K][N] row-major source): N = lane%16,
//   K = k0+8h+{0..7} then k0+16+8h+{0..7}, h = lane/16
__device__ __forceinline__ v16bf load_b(const __bf16* base, int ld, int k0, int n0) {
  int lane = threadIdx.x & 31;
  const __bf16* p = base + (size_t)(k0 + ((lane >> 4) << 3)) * ld + n0 + (lane & 15);
  v16bf r;
#pragma unroll
  for (int i = 0; i < 8; ++i) { r[i] = p[(size_t)i * ld]; r[i + 8] = p[(size_t)(i + 16) * ld]; }
  return r;
}

__device__ __forceinline__ v8f wmma_bf16(v16bf a, v16bf b, v8f c) {
  return __builtin_amdgcn_wmma_f32_16x16x32_bf16(false, a, false, b, (short)0, c,
                                                 false, false);
}

__device__ __forceinline__ float sigmf(float x) { return 1.f / (1.f + __expf(-x)); }

// ---------------------------------------------------------------------------
// Workspace layout (bytes)
// ---------------------------------------------------------------------------
constexpr size_t OFF_CNT   = 0;
constexpr size_t OFF_XN    = 256;                                    // bf16 B*T*FIN
constexpr size_t OFF_CONVW = OFF_XN    + (size_t)B_ * T_ * FIN * 2;  // bf16 384*64
constexpr size_t OFF_CONVO = OFF_CONVW + (size_t)384 * 64 * 2;       // bf16 TP*B*CC
constexpr size_t OFF_WIHT  = OFF_CONVO + (size_t)TP * B_ * CC * 2;   // bf16 64*G4
constexpr size_t OFF_WHHT  = OFF_WIHT  + (size_t)64 * G4 * 2;        // bf16 H*G4
constexpr size_t OFF_BIAS  = OFF_WHHT  + (size_t)H_ * G4 * 2;        // f32  G4
constexpr size_t OFF_HBF   = OFF_BIAS  + (size_t)G4 * 4;             // bf16 2*B*H
constexpr size_t OFF_HF32  = OFF_HBF   + (size_t)2 * B_ * H_ * 2;    // f32  B*H

// ---------------------------------------------------------------------------
// Init: reset the barrier counter and zero h(0) (ping buffer 0)
// ---------------------------------------------------------------------------
__global__ void k_init(unsigned* cnt, __bf16* hbuf) {
  int i = blockIdx.x * blockDim.x + threadIdx.x;
  if (i == 0) *cnt = 0u;
  if (i < B_ * H_) hbuf[i] = (__bf16)0.0f;
}

// ---------------------------------------------------------------------------
// L2 normalize along T per (b,f) column; emit bf16
// grid <<<B_, FIN>>>
// ---------------------------------------------------------------------------
__global__ void k_norm(const float* __restrict__ in, __bf16* __restrict__ xn) {
  int b = blockIdx.x, f = threadIdx.x;
  const float* col = in + (size_t)b * T_ * FIN + f;
  float ss = 0.f;
  for (int t = 0; t < T_; ++t) { float v = col[(size_t)t * FIN]; ss += v * v; }
  float inv = 1.f / fmaxf(sqrtf(ss), 1e-12f);
  __bf16* o = xn + (size_t)b * T_ * FIN + f;
  for (int t = 0; t < T_; ++t) o[(size_t)t * FIN] = (__bf16)(col[(size_t)t * FIN] * inv);
}

// ---------------------------------------------------------------------------
// Weight packing to bf16, [K][N] layouts for WMMA-B
// ---------------------------------------------------------------------------
__global__ void k_pack_convw(const float* __restrict__ w, __bf16* __restrict__ wp) {
  int i = blockIdx.x * 256 + threadIdx.x;            // 384*64
  if (i >= 384 * 64) return;
  int K = i >> 6, c = i & 63;
  int k = K >> 7, f = K & 127;                       // K index = k*128 + f
  wp[i] = (__bf16)w[(size_t)(c * FIN + f) * 3 + k];  // conv_w (CC,FIN,3)
}
__global__ void k_pack_wih(const float* __restrict__ w, __bf16* __restrict__ wt) {
  int i = blockIdx.x * 256 + threadIdx.x;            // 64*G4
  if (i >= 64 * G4) return;
  int k = i >> 11, g = i & (G4 - 1);
  wt[i] = (__bf16)w[(size_t)g * CC + k];             // w_ih (G4,CC)
}
__global__ void k_pack_whh(const float* __restrict__ w, __bf16* __restrict__ wt) {
  int i = blockIdx.x * 256 + threadIdx.x;            // H*G4
  if (i >= H_ * G4) return;
  int k = i >> 11, g = i & (G4 - 1);
  wt[i] = (__bf16)w[(size_t)g * H_ + k];             // w_hh (G4,H)
}
__global__ void k_bias(const float* __restrict__ bi, const float* __restrict__ bh,
                       float* __restrict__ bias) {
  int i = blockIdx.x * 256 + threadIdx.x;
  if (i < G4) bias[i] = bi[i] + bh[i];
}

// ---------------------------------------------------------------------------
// Conv1d(k=3, VALID) + bias + ReLU as implicit-im2col WMMA GEMM.
// One wave per 16(t) x 16(c) tile, K = 3*FIN = 384. Output (T',B,CC) bf16.
// ---------------------------------------------------------------------------
__global__ void k_conv(const __bf16* __restrict__ xn, const __bf16* __restrict__ wp,
                       const float* __restrict__ cb, __bf16* __restrict__ convo) {
  int wid = (blockIdx.x * blockDim.x + threadIdx.x) >> 5;
  if (wid >= 128 * 64 * 4) return;
  int ct = wid & 3, tt = (wid >> 2) & 63, b = wid >> 8;
  int t0 = tt * 16, n0 = ct * 16;
  int lane = threadIdx.x & 31;
  v8f acc = {};
#pragma unroll
  for (int kk0 = 0; kk0 < 384; kk0 += 32) {
    int k = kk0 >> 7, f0 = kk0 & 127;                // tap k fixed per 32-chunk
    int koff = (lane >> 4) << 3;
    int trow = t0 + (lane & 15) + k;
    if (trow > T_ - 1) trow = T_ - 1;                // clamp (stores masked below)
    const __bf16* p = xn + ((size_t)b * T_ + trow) * FIN + f0 + koff;
    v8bf lo = *(const v8bf*)p;
    v8bf hi = *(const v8bf*)(p + 16);
    v16bf a = make_frag(lo, hi);
    v16bf bf = load_b(wp, 64, kk0, n0);
    acc = wmma_bf16(a, bf, acc);
  }
  int c = n0 + (lane & 15);
  float biasc = cb[c];
#pragma unroll
  for (int r = 0; r < 8; ++r) {
    int t = t0 + r + ((lane >> 4) << 3);             // C/D layout: M = r (+8 hi half)
    if (t < TP) {
      float v = acc[r] + biasc;
      v = v > 0.f ? v : 0.f;
      convo[((size_t)t * B_ + b) * CC + c] = (__bf16)v;
    }
  }
}

// ---------------------------------------------------------------------------
// Persistent LSTM: 32 blocks x 256 threads (8 waves). Block owns 16 h-columns
// (and their 4 gate-column groups, w_hh slice in LDS); wave owns 16 batch rows.
// c stays in registers; h ping-pongs in global bf16; one device barrier/step.
// ---------------------------------------------------------------------------
__global__ void __launch_bounds__(256, 1)
k_lstm(const __bf16* __restrict__ convo, const __bf16* __restrict__ wihT,
       const __bf16* __restrict__ whhT, const float* __restrict__ bias,
       __bf16* __restrict__ hbuf, float* __restrict__ hout, unsigned* cnt) {
  extern __shared__ __bf16 smem[];
  __bf16* sWhh = smem;                 // [512][64]  (4 gates x 16 cols)
  __bf16* sWih = smem + 512 * 64;      // [64][64]

  const int n16  = blockIdx.x;         // 0..31  -> h cols n16*16..+15
  const int wave = threadIdx.x >> 5;   // 0..7   -> batch rows wave*16..+15
  const int lane = threadIdx.x & 31;
  const int m0   = wave * 16;

  for (int i = threadIdx.x; i < 512 * 64; i += 256) {
    int k = i >> 6, c = i & 63, g = c >> 4, n = c & 15;
    sWhh[i] = whhT[(size_t)k * G4 + g * H_ + n16 * 16 + n];
  }
  for (int i = threadIdx.x; i < 64 * 64; i += 256) {
    int k = i >> 6, c = i & 63, g = c >> 4, n = c & 15;
    sWih[i] = wihT[(size_t)k * G4 + g * H_ + n16 * 16 + n];
  }
  __syncthreads();

  const int jc = n16 * 16 + (lane & 15);
  const float bi = bias[0 * H_ + jc];
  const float bf_ = bias[1 * H_ + jc];
  const float bg = bias[2 * H_ + jc];
  const float bo = bias[3 * H_ + jc];

  v8f creg = {};
  const unsigned nb = gridDim.x;

  for (int t = 0; t < TP; ++t) {
    const __bf16* hr = hbuf + (size_t)(t & 1) * (B_ * H_);
    __bf16*       hw = hbuf + (size_t)((t + 1) & 1) * (B_ * H_);
    const __bf16* xb = convo + (size_t)t * B_ * CC;
    if (t + 1 < TP)                                  // global_prefetch_b8
      __builtin_prefetch(convo + (size_t)(t + 1) * B_ * CC + (size_t)m0 * CC, 0, 1);

    v8f a0 = {}, a1 = {}, a2 = {}, a3 = {};
#pragma unroll
    for (int k0 = 0; k0 < CC; k0 += 32) {            // x-projection, K=64
      v16bf a = load_a(xb, CC, m0, k0);
      a0 = wmma_bf16(a, load_b(sWih, 64, k0, 0),  a0);
      a1 = wmma_bf16(a, load_b(sWih, 64, k0, 16), a1);
      a2 = wmma_bf16(a, load_b(sWih, 64, k0, 32), a2);
      a3 = wmma_bf16(a, load_b(sWih, 64, k0, 48), a3);
    }
#pragma unroll 4
    for (int k0 = 0; k0 < H_; k0 += 32) {            // h @ w_hh^T, K=512
      v16bf a = load_a(hr, H_, m0, k0);
      a0 = wmma_bf16(a, load_b(sWhh, 64, k0, 0),  a0);
      a1 = wmma_bf16(a, load_b(sWhh, 64, k0, 16), a1);
      a2 = wmma_bf16(a, load_b(sWhh, 64, k0, 32), a2);
      a3 = wmma_bf16(a, load_b(sWhh, 64, k0, 48), a3);
    }
#pragma unroll
    for (int r = 0; r < 8; ++r) {
      int row = m0 + r + ((lane >> 4) << 3);
      float ig = sigmf(a0[r] + bi);
      float fg = sigmf(a1[r] + bf_);
      float gg = tanhf(a2[r] + bg);
      float og = sigmf(a3[r] + bo);
      float cv = fg * creg[r] + ig * gg;
      creg[r] = cv;
      float hv = og * tanhf(cv);
      hw[(size_t)row * H_ + jc] = (__bf16)hv;
      if (t == TP - 1) hout[(size_t)row * H_ + jc] = hv;
    }

    if (t != TP - 1) {                               // device-wide step barrier
      __threadfence();
      __syncthreads();
      if (threadIdx.x == 0) {
        __hip_atomic_fetch_add(cnt, 1u, __ATOMIC_RELEASE, __HIP_MEMORY_SCOPE_AGENT);
        unsigned target = nb * (unsigned)(t + 1);
        while (__hip_atomic_load(cnt, __ATOMIC_ACQUIRE, __HIP_MEMORY_SCOPE_AGENT) < target)
          __builtin_amdgcn_s_sleep(1);
      }
      __syncthreads();
    }
  }
}

// ---------------------------------------------------------------------------
// Final linear: out(B,10) = h @ lin_w^T + lin_b (1.3 MFLOP — scalar kernel)
// ---------------------------------------------------------------------------
__global__ void k_linear(const float* __restrict__ h, const float* __restrict__ lw,
                         const float* __restrict__ lb, float* __restrict__ out) {
  int i = blockIdx.x * blockDim.x + threadIdx.x;
  if (i >= B_ * OUTN) return;
  int b = i / OUTN, o = i % OUTN;
  float s = lb[o];
  const float* hp = h + (size_t)b * H_;
  const float* wp = lw + (size_t)o * H_;
  for (int k = 0; k < H_; ++k) s += hp[k] * wp[k];
  out[i] = s;
}

// ---------------------------------------------------------------------------
extern "C" void kernel_launch(void* const* d_in, const int* in_sizes, int n_in,
                              void* d_out, int out_size, void* d_ws, size_t ws_size,
                              hipStream_t stream) {
  (void)in_sizes; (void)n_in; (void)out_size; (void)ws_size;
  const float* input  = (const float*)d_in[0];
  const float* conv_w = (const float*)d_in[2];
  const float* conv_b = (const float*)d_in[3];
  const float* w_ih   = (const float*)d_in[4];
  const float* w_hh   = (const float*)d_in[5];
  const float* b_ih   = (const float*)d_in[6];
  const float* b_hh   = (const float*)d_in[7];
  const float* lin_w  = (const float*)d_in[8];
  const float* lin_b  = (const float*)d_in[9];
  float* out = (float*)d_out;

  char* ws = (char*)d_ws;
  unsigned* cnt   = (unsigned*)(ws + OFF_CNT);
  __bf16* xn      = (__bf16*)(ws + OFF_XN);
  __bf16* convw   = (__bf16*)(ws + OFF_CONVW);
  __bf16* convo   = (__bf16*)(ws + OFF_CONVO);
  __bf16* wihT    = (__bf16*)(ws + OFF_WIHT);
  __bf16* whhT    = (__bf16*)(ws + OFF_WHHT);
  float*  bias    = (float*)(ws + OFF_BIAS);
  __bf16* hbuf    = (__bf16*)(ws + OFF_HBF);
  float*  hf32    = (float*)(ws + OFF_HF32);

  k_init<<<(B_ * H_ + 255) / 256, 256, 0, stream>>>(cnt, hbuf);
  k_norm<<<B_, FIN, 0, stream>>>(input, xn);
  k_pack_convw<<<(384 * 64 + 255) / 256, 256, 0, stream>>>(conv_w, convw);
  k_pack_wih<<<(64 * G4 + 255) / 256, 256, 0, stream>>>(w_ih, wihT);
  k_pack_whh<<<(H_ * G4 + 255) / 256, 256, 0, stream>>>(w_hh, whhT);
  k_bias<<<(G4 + 255) / 256, 256, 0, stream>>>(b_ih, b_hh, bias);
  k_conv<<<4096, 256, 0, stream>>>(xn, convw, conv_b, convo);
  size_t lds = (size_t)(512 * 64 + 64 * 64) * sizeof(__bf16);   // 73728 B
  k_lstm<<<32, 256, lds, stream>>>(convo, wihT, whhT, bias, hbuf, hf32, cnt);
  k_linear<<<(B_ * OUTN + 255) / 256, 256, 0, stream>>>(hf32, lin_w, lin_b, out);
}
